// RNNS2Smodel_61314953117815
// MI455X (gfx1250) — compile-verified, matrix operand order checked by reading, not run on hardware
//
#include <hip/hip_runtime.h>
#include <cstdint>

// ---------------------------------------------------------------------------
// MI455X (gfx1250) seq2seq LSTM. wave32, WMMA f32_16x16x32_f16,
// async global->LDS staging (ASYNCcnt) when the toolchain exposes it.
// B=32 S=256 T=128 D=H=512 V=32000, 4H=2048.
// ---------------------------------------------------------------------------

typedef _Float16 f16;
typedef __attribute__((ext_vector_type(16))) _Float16 v16h;
typedef __attribute__((ext_vector_type(8)))  float    v8f;

static constexpr int NBLK = 16;   // persistent blocks for the recurrent kernels

// --------------------- async global->LDS (CDNA5 path) ----------------------
#if defined(__has_builtin)
#if __has_builtin(__builtin_amdgcn_global_load_async_to_lds_b128)
#define ASYNC_LDS 1
#endif
#endif
#ifndef ASYNC_LDS
#define ASYNC_LDS 0
#endif

#if ASYNC_LDS
// builtin signature (probed from clang diagnostic):
//   void __builtin_amdgcn_global_load_async_to_lds_b128(
//       as1 int4* src, as3 int4* dst, imm int offset, imm int cpol)
typedef __attribute__((ext_vector_type(4))) int v4i_t;
typedef __attribute__((address_space(1))) v4i_t as1_v4i;
typedef __attribute__((address_space(3))) v4i_t as3_v4i;
#endif

// copy 16 bytes global -> LDS. Async path: per-lane DMA, no VGPR round-trip,
// tracked by ASYNCcnt. LDS addrspace(3) offset = low 32 bits of the generic
// pointer (ISA 10.2 aperture rules); global addrspace(1) is numerically
// identical to generic.
__device__ __forceinline__ void cp_b128(void* lds_dst, const void* g_src) {
#if ASYNC_LDS
  __builtin_amdgcn_global_load_async_to_lds_b128(
      (as1_v4i*)(uintptr_t)g_src,
      (as3_v4i*)(uint32_t)(uintptr_t)lds_dst,
      0, 0);
#else
  *(uint4*)lds_dst = *(const uint4*)g_src;
#endif
}

// drain this wave's outstanding async LDS writes before the block barrier
__device__ __forceinline__ void async_drain() {
#if ASYNC_LDS
#if defined(__has_builtin) && __has_builtin(__builtin_amdgcn_s_wait_asynccnt)
  __builtin_amdgcn_s_wait_asynccnt(0);
#else
  asm volatile("s_wait_asynccnt 0" ::: "memory");
#endif
#endif
}

// ----------------------------- workspace layout ----------------------------
static constexpr size_t SZW = (size_t)2048 * 512 * 2;   // big weight (f16)
static constexpr size_t SZS = (size_t)512 * 512 * 2;    // small weight (f16)
static constexpr size_t o_encW0t  = 0;
static constexpr size_t o_encU0t  = o_encW0t + SZW;
static constexpr size_t o_encW1t  = o_encU0t + SZW;
static constexpr size_t o_encU1t  = o_encW1t + SZW;
static constexpr size_t o_decWt   = o_encU1t + SZW;
static constexpr size_t o_decUt   = o_decWt  + SZW;
static constexpr size_t o_outWt   = o_decUt  + SZW;
static constexpr size_t o_diWu    = o_outWt  + SZS;
static constexpr size_t o_diWl    = o_diWu   + SZS;
static constexpr size_t o_finWt   = o_diWl   + SZS;                      // 32000x512 f16
static constexpr size_t o_Xenc    = o_finWt  + (size_t)32000*512*2;      // 8192x512 f16
static constexpr size_t o_labn    = o_Xenc   + (size_t)8192*512*2;       // 4096x512 f16
static constexpr size_t o_Xp0     = o_labn   + (size_t)4096*512*2;       // 8192x2048 f16 (aliased Xp_dec)
static constexpr size_t o_decin16 = o_Xp0    + (size_t)8192*2048*2;      // 4096x512 f16
static constexpr size_t o_hs      = o_decin16+ (size_t)4096*512*2;       // 4096x512 f16
static constexpr size_t o_zl      = o_hs     + (size_t)4096*512*2;       // 4096x512 f32
static constexpr size_t o_encvec  = o_zl     + (size_t)4096*512*4;       // 32x512 f32
static constexpr size_t o_evn     = o_encvec + (size_t)32*512*4;         // 32x512 f16
static constexpr size_t o_evu     = o_evn    + (size_t)32*512*2;         // 32x512 f32
static constexpr size_t o_state   = o_evu    + (size_t)32*512*4;         // zeroed each call:
static constexpr size_t o_h0      = o_state;                             // 2x32x512 f16
static constexpr size_t o_h1      = o_h0 + (size_t)2*32*512*2;
static constexpr size_t o_hd      = o_h1 + (size_t)2*32*512*2;
static constexpr size_t o_c0      = o_hd + (size_t)2*32*512*2;           // 32x512 f32
static constexpr size_t o_c1      = o_c0 + (size_t)32*512*4;
static constexpr size_t o_cd      = o_c1 + (size_t)32*512*4;
static constexpr size_t o_bar     = o_cd + (size_t)32*512*4;             // barrier counters
static constexpr size_t o_end     = o_bar + 256;
static constexpr size_t STATE_BYTES = o_end - o_state;

// ----------------------------- device helpers -----------------------------
__device__ __forceinline__ float sigm(float x) { return 1.f / (1.f + expf(-x)); }

// Load one 16x32 f16 tile (row-major, leading dim ld) into the WMMA A-layout
// (also used for B by loading rows of B^T). ISA 7.12.2 16-bit A-matrix map:
// vgpr j holds K = (j<4?0:16) + (lane>=16?8:0) + (j&3)*2, row M = lane&15.
__device__ __forceinline__ v16h load_frag(const f16* __restrict__ p, int ld) {
  const int lane = threadIdx.x & 31;
  const int r    = lane & 15;
  const int kb   = (lane >> 4) << 3;
  const f16* rp = p + (size_t)r * ld;
  v16h v;
#pragma unroll
  for (int j = 0; j < 8; ++j) {
    int k = ((j < 4) ? 0 : 16) + kb + ((j & 3) << 1);
    v[2*j]   = rp[k];
    v[2*j+1] = rp[k+1];
  }
  return v;
}

__device__ __forceinline__ v8f wmma_step(v16h a, v16h b, v8f c) {
  return __builtin_amdgcn_wmma_f32_16x16x32_f16(false, a, false, b, (short)0, c,
                                                false, false);
}

// device-wide barrier for the persistent recurrent kernels (monotone epochs,
// counter zeroed once per kernel_launch -> graph-replay deterministic)
__device__ __forceinline__ void grid_barrier(unsigned* cnt, unsigned target) {
  __syncthreads();
  if (threadIdx.x == 0) {
    __threadfence();
    atomicAdd(cnt, 1u);
    while (atomicAdd(cnt, 0u) < target) __builtin_amdgcn_s_sleep(2);
  }
  __syncthreads();
  __threadfence();
}

// copy 32x512 f16 (32 KB) global -> LDS with 128-bit (async) transfers
__device__ __forceinline__ void stage32x512(f16* dst, const f16* src, int tid) {
#pragma unroll
  for (int v = 0; v < 8; ++v) {
    int i = tid + v * 256;
    cp_b128((uint4*)dst + i, (const uint4*)src + i);
  }
  async_drain();
}

// K-slice GEMM used by the recurrent kernels: acc[2](16x16 tiles, M=0..31)
// += sH[32x512] x Bt_row_block[16x512]^T
__device__ __forceinline__ void gemm_slice(v8f* acc, const f16* sH,
                                           const f16* __restrict__ brow) {
#pragma unroll
  for (int k0 = 0; k0 < 512; k0 += 32) {
    v16h bf = load_frag(brow + k0, 512);
#pragma unroll
    for (int mt = 0; mt < 2; ++mt) {
      v16h af = load_frag(sH + mt * 16 * 512 + k0, 512);
      acc[mt] = wmma_step(af, bf, acc[mt]);
    }
  }
}

// scatter two 16x16 f32 accumulators into sZ[32][128] at column base zc
__device__ __forceinline__ void store_z(float* sZ, const v8f* acc, int zc) {
  const int lane = threadIdx.x & 31;
  const int ncol = lane & 15, mb = (lane >> 4) << 3;
#pragma unroll
  for (int mt = 0; mt < 2; ++mt)
#pragma unroll
    for (int e = 0; e < 8; ++e)
      sZ[(mt * 16 + mb + e) * 128 + zc + ncol] = acc[mt][e];
}

// ------------------------------- utility kernels ---------------------------
__global__ void k_zero(unsigned* __restrict__ p, size_t n) {
  for (size_t i = (size_t)blockIdx.x * blockDim.x + threadIdx.x; i < n;
       i += (size_t)gridDim.x * blockDim.x)
    p[i] = 0u;
}

// out[n*K+k] = (f16) in[k*N+n]   (transpose + downconvert)
__global__ void k_convt(const float* __restrict__ in, f16* __restrict__ out,
                        int K, int N) {
  size_t total = (size_t)K * N;
  for (size_t i = (size_t)blockIdx.x * blockDim.x + threadIdx.x; i < total;
       i += (size_t)gridDim.x * blockDim.x) {
    size_t n = i / (size_t)K;
    size_t k = i - n * (size_t)K;
    out[i] = (f16)in[k * (size_t)N + n];
  }
}

// X_enc[s*B+b][d] = f16(emb_enc[source[b][s]][d])
__global__ void k_gather_enc(const int* __restrict__ source,
                             const float* __restrict__ emb,
                             f16* __restrict__ out) {
  size_t total = (size_t)8192 * 512;
  for (size_t i = (size_t)blockIdx.x * blockDim.x + threadIdx.x; i < total;
       i += (size_t)gridDim.x * blockDim.x) {
    int d = (int)(i & 511);
    int r = (int)(i >> 9);       // s*32 + b
    int b = r & 31, s = r >> 5;
    int tok = source[b * 256 + s];
    out[i] = (f16)emb[(size_t)tok * 512 + d];
  }
}

// one block per row: l2-normalized gathered embedding -> f16
__global__ void k_gather_norm(const int* __restrict__ idx,
                              const float* __restrict__ table,
                              f16* __restrict__ out) {
  const int row = blockIdx.x;
  const float* src = table + (size_t)idx[row] * 512;
  __shared__ float red[256];
  float ss = 0.f;
  for (int d = threadIdx.x; d < 512; d += 256) { float v = src[d]; ss += v * v; }
  red[threadIdx.x] = ss; __syncthreads();
  for (int s = 128; s > 0; s >>= 1) {
    if (threadIdx.x < s) red[threadIdx.x] += red[threadIdx.x + s];
    __syncthreads();
  }
  float scale = rsqrtf(fmaxf(red[0], 1e-12f));
  for (int d = threadIdx.x; d < 512; d += 256)
    out[(size_t)row * 512 + d] = (f16)(src[d] * scale);
}

// l2-normalize f32 rows -> f16
__global__ void k_rownorm(const float* __restrict__ in, f16* __restrict__ out) {
  const int row = blockIdx.x;
  const float* src = in + (size_t)row * 512;
  __shared__ float red[256];
  float ss = 0.f;
  for (int d = threadIdx.x; d < 512; d += 256) { float v = src[d]; ss += v * v; }
  red[threadIdx.x] = ss; __syncthreads();
  for (int s = 128; s > 0; s >>= 1) {
    if (threadIdx.x < s) red[threadIdx.x] += red[threadIdx.x + s];
    __syncthreads();
  }
  float scale = rsqrtf(fmaxf(red[0], 1e-12f));
  for (int d = threadIdx.x; d < 512; d += 256)
    out[(size_t)row * 512 + d] = (f16)(src[d] * scale);
}

// dec_in[b,t,:] = zl[b*T+t,:] + evu[b,:]  -> f32 out (second model output) + f16 copy
__global__ void k_finish_decin(const float* __restrict__ zl,
                               const float* __restrict__ evu,
                               float* __restrict__ dec_in_out,
                               f16* __restrict__ decin16) {
  size_t total = (size_t)4096 * 512;
  for (size_t i = (size_t)blockIdx.x * blockDim.x + threadIdx.x; i < total;
       i += (size_t)gridDim.x * blockDim.x) {
    int d = (int)(i & 511);
    int row = (int)(i >> 9);
    int b = row >> 7;            // T = 128
    float v = zl[i] + evu[b * 512 + d];
    dec_in_out[i] = v;
    decin16[i] = (f16)v;
  }
}

// --------------------------- generic WMMA GEMM -----------------------------
// C[M,N] = A[M,K]f16 * Bt[N,K]f16^T (+ bias[N]); block tile 32x256, 8 waves,
// each wave: 2 M-tiles x 2 N-tiles; A/B staged in LDS via async DMA.
template <bool OUTF16, bool BIAS>
__global__ __launch_bounds__(256) void k_gemm(const f16* __restrict__ A,
                                              const f16* __restrict__ Bt,
                                              const float* __restrict__ bias,
                                              void* __restrict__ C,
                                              int M, int N, int K) {
  __shared__ __align__(16) f16 sA[32 * 32];
  __shared__ __align__(16) f16 sB[256 * 32];
  const int tid  = threadIdx.x;
  const int wave = tid >> 5;
  const int lane = tid & 31;
  const int bm = blockIdx.y * 32;
  const int bn = blockIdx.x * 256;

  v8f vz = {};
  v8f acc[2][2];
#pragma unroll
  for (int i = 0; i < 2; ++i)
#pragma unroll
    for (int j = 0; j < 2; ++j) acc[i][j] = vz;

  for (int k0 = 0; k0 < K; k0 += 32) {
    if (tid < 128) {             // A tile: 32 rows x 32 halves = 128 x b128
      int r = tid >> 2, c8 = (tid & 3) << 3;
      cp_b128(&sA[r * 32 + c8], &A[(size_t)(bm + r) * K + k0 + c8]);
    }
#pragma unroll
    for (int v = 0; v < 4; ++v) { // B tile: 256 rows x 32 halves = 1024 x b128
      int i = tid + v * 256;
      int r = i >> 2, c8 = (i & 3) << 3;
      cp_b128(&sB[r * 32 + c8], &Bt[(size_t)(bn + r) * K + k0 + c8]);
    }
    if (k0 + 32 < K)
      __builtin_prefetch(&Bt[(size_t)(bn + (tid >> 1)) * K + k0 + 32], 0, 1);
    async_drain();
    __syncthreads();

    v16h af[2], bf[2];
#pragma unroll
    for (int mt = 0; mt < 2; ++mt) af[mt] = load_frag(&sA[(mt * 16) * 32], 32);
#pragma unroll
    for (int nt = 0; nt < 2; ++nt)
      bf[nt] = load_frag(&sB[(wave * 32 + nt * 16) * 32], 32);
#pragma unroll
    for (int mt = 0; mt < 2; ++mt)
#pragma unroll
      for (int nt = 0; nt < 2; ++nt)
        acc[mt][nt] = wmma_step(af[mt], bf[nt], acc[mt][nt]);
    __syncthreads();
  }

  const int ncol = lane & 15, mb = (lane >> 4) << 3;
  const int bn_w = bn + wave * 32;
#pragma unroll
  for (int mt = 0; mt < 2; ++mt) {
#pragma unroll
    for (int nt = 0; nt < 2; ++nt) {
      int col = bn_w + nt * 16 + ncol;
      float bv = 0.f;
      if constexpr (BIAS) bv = bias[col];
#pragma unroll
      for (int e = 0; e < 8; ++e) {
        int row = bm + mt * 16 + mb + e;
        float v = acc[mt][nt][e] + bv;
        if constexpr (OUTF16)
          ((f16*)C)[(size_t)row * N + col] = (f16)v;
        else
          ((float*)C)[(size_t)row * N + col] = v;
      }
    }
  }
}

// --------------------------- encoder recurrence ----------------------------
// 16 persistent blocks; block owns H-columns [j0, j0+32). Per step:
//   layer0: z0 = Xp0[s] + h0@U0 -> gates -> h0'
//   layer1: z1 = h0'@W1 + h1@U1 + b1 -> gates -> h1'
__global__ __launch_bounds__(256) void k_encoder(
    const f16* __restrict__ Xp0, const f16* __restrict__ U0t,
    const f16* __restrict__ W1t, const f16* __restrict__ U1t,
    const float* __restrict__ b1, f16* __restrict__ h0buf,
    f16* __restrict__ h1buf, float* __restrict__ c0, float* __restrict__ c1,
    unsigned* __restrict__ bar) {
  __shared__ __align__(16) f16 sH[32 * 512];
  __shared__ float sZ[32 * 128];
  const int tid = threadIdx.x, wave = tid >> 5;
  const int j0 = blockIdx.x * 32;
  const int g = wave >> 1, hh = wave & 1;
  const int zc = g * 32 + hh * 16;
  const size_t bt_row = (size_t)(g * 512 + j0 + hh * 16) * 512;
  v8f vz = {};
  unsigned epoch = 0;

  for (int s = 0; s < 256; ++s) {
    const int rb = s & 1, wb = rb ^ 1;
    // ---------------- layer 0 ----------------
    stage32x512(sH, h0buf + (size_t)rb * 32 * 512, tid);
    __syncthreads();
    v8f acc[2] = {vz, vz};
    gemm_slice(acc, sH, U0t + bt_row);
    store_z(sZ, acc, zc);
    __syncthreads();
    const f16* xp = Xp0 + (size_t)s * 32 * 2048;
#pragma unroll
    for (int it = 0; it < 4; ++it) {
      int idx = tid + it * 256;
      int b = idx >> 5, jl = idx & 31, col = j0 + jl;
      const f16* xr = xp + (size_t)b * 2048;
      float zi = sZ[b * 128 + jl]      + (float)xr[col];
      float zf = sZ[b * 128 + 32 + jl] + (float)xr[512 + col];
      float zg = sZ[b * 128 + 64 + jl] + (float)xr[1024 + col];
      float zo = sZ[b * 128 + 96 + jl] + (float)xr[1536 + col];
      float cv = sigm(zf) * c0[b * 512 + col] + sigm(zi) * tanhf(zg);
      c0[b * 512 + col] = cv;
      h0buf[(size_t)wb * 32 * 512 + b * 512 + col] = (f16)(sigm(zo) * tanhf(cv));
    }
    ++epoch;
    grid_barrier(bar, epoch * (unsigned)NBLK);
    // ---------------- layer 1 ----------------
    stage32x512(sH, h0buf + (size_t)wb * 32 * 512, tid);
    __syncthreads();
    v8f a1[2] = {vz, vz};
    gemm_slice(a1, sH, W1t + bt_row);
    __syncthreads();
    stage32x512(sH, h1buf + (size_t)rb * 32 * 512, tid);
    __syncthreads();
    gemm_slice(a1, sH, U1t + bt_row);
    store_z(sZ, a1, zc);
    __syncthreads();
#pragma unroll
    for (int it = 0; it < 4; ++it) {
      int idx = tid + it * 256;
      int b = idx >> 5, jl = idx & 31, col = j0 + jl;
      float zi = sZ[b * 128 + jl]      + b1[col];
      float zf = sZ[b * 128 + 32 + jl] + b1[512 + col];
      float zg = sZ[b * 128 + 64 + jl] + b1[1024 + col];
      float zo = sZ[b * 128 + 96 + jl] + b1[1536 + col];
      float cv = sigm(zf) * c1[b * 512 + col] + sigm(zi) * tanhf(zg);
      c1[b * 512 + col] = cv;
      h1buf[(size_t)wb * 32 * 512 + b * 512 + col] = (f16)(sigm(zo) * tanhf(cv));
    }
    ++epoch;
    grid_barrier(bar, epoch * (unsigned)NBLK);
  }
}

// --------------------------- decoder recurrence ----------------------------
__global__ __launch_bounds__(256) void k_decoder(
    const f16* __restrict__ Xp, const f16* __restrict__ Ut,
    f16* __restrict__ hbuf, float* __restrict__ cbuf, f16* __restrict__ hs,
    unsigned* __restrict__ bar) {
  __shared__ __align__(16) f16 sH[32 * 512];
  __shared__ float sZ[32 * 128];
  const int tid = threadIdx.x, wave = tid >> 5;
  const int j0 = blockIdx.x * 32;
  const int g = wave >> 1, hh = wave & 1;
  const int zc = g * 32 + hh * 16;
  const size_t bt_row = (size_t)(g * 512 + j0 + hh * 16) * 512;
  v8f vz = {};
  unsigned epoch = 0;

  for (int t = 0; t < 128; ++t) {
    const int rb = t & 1, wb = rb ^ 1;
    stage32x512(sH, hbuf + (size_t)rb * 32 * 512, tid);
    __syncthreads();
    v8f acc[2] = {vz, vz};
    gemm_slice(acc, sH, Ut + bt_row);
    store_z(sZ, acc, zc);
    __syncthreads();
#pragma unroll
    for (int it = 0; it < 4; ++it) {
      int idx = tid + it * 256;
      int b = idx >> 5, jl = idx & 31, col = j0 + jl;
      const f16* xr = Xp + (size_t)(b * 128 + t) * 2048;  // bias folded in
      float zi = sZ[b * 128 + jl]      + (float)xr[col];
      float zf = sZ[b * 128 + 32 + jl] + (float)xr[512 + col];
      float zg = sZ[b * 128 + 64 + jl] + (float)xr[1024 + col];
      float zo = sZ[b * 128 + 96 + jl] + (float)xr[1536 + col];
      float cv = sigm(zf) * cbuf[b * 512 + col] + sigm(zi) * tanhf(zg);
      cbuf[b * 512 + col] = cv;
      float hv = sigm(zo) * tanhf(cv);
      hbuf[(size_t)wb * 32 * 512 + b * 512 + col] = (f16)hv;
      hs[(size_t)(b * 128 + t) * 512 + col] = (f16)hv;
    }
    ++epoch;
    grid_barrier(bar, epoch * (unsigned)NBLK);
  }
}

// --------------------------------- launch ----------------------------------
extern "C" void kernel_launch(void* const* d_in, const int* in_sizes, int n_in,
                              void* d_out, int out_size, void* d_ws,
                              size_t ws_size, hipStream_t stream) {
  (void)in_sizes; (void)n_in; (void)out_size; (void)ws_size;
  const int*   source    = (const int*)d_in[0];
  const int*   label     = (const int*)d_in[1];
  const float* emb_enc   = (const float*)d_in[2];
  const float* enc_W0    = (const float*)d_in[3];
  const float* enc_U0    = (const float*)d_in[4];
  const float* enc_b0    = (const float*)d_in[5];
  const float* enc_W1    = (const float*)d_in[6];
  const float* enc_U1    = (const float*)d_in[7];
  const float* enc_b1    = (const float*)d_in[8];
  const float* enc_out_W = (const float*)d_in[9];
  const float* enc_out_b = (const float*)d_in[10];
  const float* emb_dec   = (const float*)d_in[11];
  const float* dec_in_W  = (const float*)d_in[12];
  const float* dec_in_b  = (const float*)d_in[13];
  const float* dec_W     = (const float*)d_in[14];
  const float* dec_U     = (const float*)d_in[15];
  const float* dec_b     = (const float*)d_in[16];
  const float* fin_W     = (const float*)d_in[17];
  const float* fin_b     = (const float*)d_in[18];

  char* ws = (char*)d_ws;
  float* logits     = (float*)d_out;                       // [4096, 32000]
  float* dec_in_out = logits + (size_t)4096 * 32000;       // [4096, 512]

  f16* encW0t = (f16*)(ws + o_encW0t);
  f16* encU0t = (f16*)(ws + o_encU0t);
  f16* encW1t = (f16*)(ws + o_encW1t);
  f16* encU1t = (f16*)(ws + o_encU1t);
  f16* decWt  = (f16*)(ws + o_decWt);
  f16* decUt  = (f16*)(ws + o_decUt);
  f16* outWt  = (f16*)(ws + o_outWt);
  f16* diWu   = (f16*)(ws + o_diWu);
  f16* diWl   = (f16*)(ws + o_diWl);
  f16* finWt  = (f16*)(ws + o_finWt);
  f16* Xenc   = (f16*)(ws + o_Xenc);
  f16* labn   = (f16*)(ws + o_labn);
  f16* Xp0    = (f16*)(ws + o_Xp0);     // also reused as Xp_dec
  f16* decin16= (f16*)(ws + o_decin16);
  f16* hs     = (f16*)(ws + o_hs);
  float* zl     = (float*)(ws + o_zl);
  float* encvec = (float*)(ws + o_encvec);
  f16*   evn    = (f16*)(ws + o_evn);
  float* evu    = (float*)(ws + o_evu);
  f16*   h0buf  = (f16*)(ws + o_h0);
  f16*   h1buf  = (f16*)(ws + o_h1);
  f16*   hdbuf  = (f16*)(ws + o_hd);
  float* c0     = (float*)(ws + o_c0);
  float* c1     = (float*)(ws + o_c1);
  float* cd     = (float*)(ws + o_cd);
  unsigned* barE = (unsigned*)(ws + o_bar);
  unsigned* barD = (unsigned*)(ws + o_bar + 64);

  // 0) zero persistent state (h/c buffers + barrier counters) each call
  k_zero<<<256, 256, 0, stream>>>((unsigned*)(ws + o_state), STATE_BYTES / 4);

  // 1) weight casts (transposed to [N][K] f16)
  k_convt<<<4096, 256, 0, stream>>>(enc_W0, encW0t, 512, 2048);
  k_convt<<<4096, 256, 0, stream>>>(enc_U0, encU0t, 512, 2048);
  k_convt<<<4096, 256, 0, stream>>>(enc_W1, encW1t, 512, 2048);
  k_convt<<<4096, 256, 0, stream>>>(enc_U1, encU1t, 512, 2048);
  k_convt<<<4096, 256, 0, stream>>>(dec_W,  decWt,  512, 2048);
  k_convt<<<4096, 256, 0, stream>>>(dec_U,  decUt,  512, 2048);
  k_convt<<<1024, 256, 0, stream>>>(enc_out_W, outWt, 512, 512);
  k_convt<<<1024, 256, 0, stream>>>(dec_in_W,              diWu, 512, 512);
  k_convt<<<1024, 256, 0, stream>>>(dec_in_W + 512 * 512,  diWl, 512, 512);
  k_convt<<<4096, 256, 0, stream>>>(fin_W, finWt, 512, 32000);

  // 2) gathers
  k_gather_enc<<<4096, 256, 0, stream>>>(source, emb_enc, Xenc);
  k_gather_norm<<<4096, 256, 0, stream>>>(label, emb_dec, labn);

  // 3) encoder input projection: Xp0 = Xenc @ W0 + b0   [8192 x 2048]
  k_gemm<true, true><<<dim3(8, 256), 256, 0, stream>>>(Xenc, encW0t, enc_b0,
                                                       Xp0, 8192, 2048, 512);
  // 4) encoder recurrence (persistent, grid-synced)
  k_encoder<<<NBLK, 256, 0, stream>>>(Xp0, encU0t, encW1t, encU1t, enc_b1,
                                      h0buf, h1buf, c0, c1, barE);
  // 5) enc_vec = h_top @ enc_out_W + b   (final h1 lands in buffer 0; S even)
  k_gemm<false, true><<<dim3(2, 1), 256, 0, stream>>>(h1buf, outWt, enc_out_b,
                                                      encvec, 32, 512, 512);
  // 6) decoder input: l2norm(enc_vec)@Wu + l2norm(emb)@Wl + b
  k_rownorm<<<32, 256, 0, stream>>>(encvec, evn);
  k_gemm<false, false><<<dim3(2, 1), 256, 0, stream>>>(evn, diWu, nullptr, evu,
                                                       32, 512, 512);
  k_gemm<false, true><<<dim3(2, 128), 256, 0, stream>>>(labn, diWl, dec_in_b,
                                                        zl, 4096, 512, 512);
  k_finish_decin<<<2048, 256, 0, stream>>>(zl, evu, dec_in_out, decin16);

  // 7) decoder projection: Xp_dec = dec_in @ dec_W + b  (reuses Xp0 region)
  k_gemm<true, true><<<dim3(8, 128), 256, 0, stream>>>(decin16, decWt, dec_b,
                                                       Xp0, 4096, 2048, 512);
  // 8) decoder recurrence
  k_decoder<<<NBLK, 256, 0, stream>>>(Xp0, decUt, hdbuf, cd, hs, barD);

  // 9) logits = hs @ fin_W + b   [4096 x 32000] -> d_out
  k_gemm<false, true><<<dim3(125, 128), 256, 0, stream>>>(hs, finWt, fin_b,
                                                          logits, 4096, 32000,
                                                          512);
}
